// NodeNet_39170101740082
// MI455X (gfx1250) — matrix operand order repaired and checked by reading, not compiled, not run on hardware
//
#include <hip/hip_runtime.h>
#include <hip/hip_bf16.h>

typedef unsigned short u16;
typedef __attribute__((ext_vector_type(16))) __bf16 v16bf;
typedef __attribute__((ext_vector_type(8)))  float  v8f;

#define HB 2048   // degree histogram bins

// ---------- helpers ----------
__device__ inline u16 f2bf(float f) {
    unsigned int u = __float_as_uint(f);
    unsigned int r = u + 0x7fffu + ((u >> 16) & 1u);
    return (u16)(r >> 16);
}

union Frag { v16bf v; uint4 q[2]; };

// ---------- fills ----------
__global__ void fill_i32_kernel(int* p, int v, int n) {
    int i = blockIdx.x * 256 + threadIdx.x;
    if (i < n) p[i] = v;
}
__global__ void fill_f32_kernel(float* p, float v, int n) {
    int i = blockIdx.x * 256 + threadIdx.x;
    if (i < n) p[i] = v;
}

// ---------- graph statistics ----------
__global__ void count_deg_kernel(const int* __restrict__ ei, int* deg_src, int* indeg, int E) {
    int e = blockIdx.x * 256 + threadIdx.x;
    if (e < E) {
        atomicAdd(deg_src + ei[e], 1);      // out-degree at src (struct feats)
        atomicAdd(indeg  + ei[E + e], 1);   // in-degree at dst (gcn/sage)
    }
}
__global__ void hist_kernel(const int* __restrict__ deg, int* hist, int N) {
    int i = blockIdx.x * 256 + threadIdx.x;
    if (i < N) {
        int d = deg[i]; if (d > HB - 1) d = HB - 1;
        atomicAdd(hist + d, 1);
    }
}
__global__ void median_kernel(const int* __restrict__ hist, float* med, int target) {
    if (threadIdx.x == 0 && blockIdx.x == 0) {
        int cum = 0;
        for (int i = 0; i < HB; ++i) {
            cum += hist[i];
            if (cum > target) { *med = (float)i; return; }
        }
        *med = (float)(HB - 1);
    }
}
__global__ void invdeg_kernel(const int* __restrict__ indeg, float* dinv, float* cntinv, int N) {
    int i = blockIdx.x * 256 + threadIdx.x;
    if (i < N) {
        float di = (float)indeg[i];
        dinv[i]   = rsqrtf(di + 1.0f);          // GCN deg includes self loop
        cntinv[i] = 1.0f / fmaxf(di, 1.0f);     // SAGE mean divisor
    }
}

// ---------- xin = [normalize(x) | struct_feats | pad] as bf16 [N,160] ----------
__global__ __launch_bounds__(256) void xin_kernel(
    const float* __restrict__ x, const int* __restrict__ deg_src,
    const float* __restrict__ med, u16* __restrict__ xin, int N, float degMeanInv)
{
    int node = blockIdx.x * 8 + (threadIdx.x >> 5);
    if (node >= N) return;
    int lane = threadIdx.x & 31;
    const float4 v = *(const float4*)(x + (long)node * 128 + lane * 4);
    float s2 = v.x * v.x + v.y * v.y + v.z * v.z + v.w * v.w;
#pragma unroll
    for (int off = 16; off >= 1; off >>= 1) s2 += __shfl_xor(s2, off, 32);
    float inv = 1.0f / fmaxf(sqrtf(s2), 1e-12f);
    u16* xp = xin + (long)node * 160;
    u16* op = xp + lane * 4;
    op[0] = f2bf(v.x * inv); op[1] = f2bf(v.y * inv);
    op[2] = f2bf(v.z * inv); op[3] = f2bf(v.w * inv);
    if (lane < 8) {                       // cols 128..159: 4 feats + zero pad
        float f0 = 0.f, f1 = 0.f, f2 = 0.f, f3 = 0.f;
        if (lane == 0) {
            float d = (float)deg_src[node];
            f0 = d * degMeanInv;
            f1 = log1pf(d);
            f2 = rsqrtf(fmaxf(d, 1.0f));
            f3 = (d > *med) ? 1.0f : 0.0f;
        }
        u16* fp = xp + 128 + lane * 4;
        fp[0] = f2bf(f0); fp[1] = f2bf(f1); fp[2] = f2bf(f2); fp[3] = f2bf(f3);
    }
}

// ---------- weight convert f32 [K,Hout] -> bf16 [Kpad,Hout] (zero pad rows) ----------
__global__ void wconv_kernel(const float* __restrict__ src, u16* __restrict__ dst,
                             int K, int Hout, int Kpad)
{
    int idx = blockIdx.x * 256 + threadIdx.x;
    int total = Kpad * Hout;
    if (idx < total) {
        int i = idx / Hout, j = idx - i * Hout;
        dst[idx] = (i < K) ? f2bf(src[i * Hout + j]) : (u16)0;
    }
}

// ---------- f32 [N,128] -> bf16 at dst (strided rows, for JK concat slices) ----------
__global__ void cvt_rows_kernel(const float* __restrict__ src, u16* __restrict__ dst,
                                int dstStride, int N)
{
    int idx = blockIdx.x * 256 + threadIdx.x;
    if (idx < N * 128) {
        int row = idx >> 7, col = idx & 127;
        dst[(long)row * dstStride + col] = f2bf(src[idx]);
    }
}

// ---------- WMMA GEMM: C[N,HOUT] (f32) = A(bf16,[N,K],lda) @ B(bf16,[K,HOUT]) ----------
// flags: bit0 = accumulate into existing C, bit1 = fused ReLU
template <int NCT>
__global__ __launch_bounds__(256) void wmma_gemm_kernel(
    const u16* __restrict__ A, int lda,
    const u16* __restrict__ B,
    const float* __restrict__ bias,
    float* __restrict__ C, int ldc,
    int Nrows, int K, int flags)
{
    constexpr int HOUT = NCT * 16;
    extern __shared__ u16 smem[];                  // [K x HOUT] bf16 weight tile
    const int tid = threadIdx.x;
    {
        // Async DMA of the weight tile straight into LDS (no VGPR round-trip).
        // K*HOUT/8 is always a multiple of 256 -> EXEC uniform for every issue.
        const int n4 = (K * HOUT) / 8;             // 16-byte chunks
        for (int i = tid; i < n4; i += 256) {
            unsigned int lds_addr = (unsigned int)(size_t)(smem + i * 8);
            const u16* gsrc = B + i * 8;
            asm volatile("global_load_async_to_lds_b128 %0, %1, off"
                         :: "v"(lds_addr), "v"(gsrc) : "memory");
        }
        asm volatile("s_wait_asynccnt 0" ::: "memory");
    }
    __syncthreads();

    const int wave = tid >> 5, lane = tid & 31;
    const int m = lane & 15, hlf = lane >> 4;
    const int rowBase = (blockIdx.x * 8 + wave) * 16;
    int arow = rowBase + m; if (arow >= Nrows) arow = Nrows - 1;
    const u16* aptr = A + (long)arow * lda;

    v8f acc[NCT] = {};
    for (int kc = 0; kc < K; kc += 32) {
        // A fragment: 16-bit 16x32 layout — lane holds k in [b,b+8)∪[16+b,16+b+8), b=8*hlf
        Frag a;
        const u16* ap = aptr + kc + hlf * 8;
        a.q[0] = *(const uint4*)(ap);
        a.q[1] = *(const uint4*)(ap + 16);
        __builtin_prefetch((const void*)(aptr + kc + 64), 0, 3);   // global_prefetch_b8, near
        const u16* bp = smem + (kc + hlf * 8) * HOUT + m;  // B fragment: col n = lane&15, k in [b,b+8)∪[16+b,16+b+8), b=8*hlf
#pragma unroll
        for (int c = 0; c < NCT; ++c) {
            Frag b;
#pragma unroll
            for (int s = 0; s < 16; ++s)
                ((u16*)&b)[s] = bp[(s + (s >= 8 ? 8 : 0)) * HOUT + c * 16];
            acc[c] = __builtin_amdgcn_wmma_f32_16x16x32_bf16(
                false, a.v, false, b.v, (short)0, acc[c], false, false);
        }
    }

    const bool accum = flags & 1, relu = flags & 2;
#pragma unroll
    for (int c = 0; c < NCT; ++c) {
        int col = c * 16 + m;
        float bv = bias ? bias[col] : 0.0f;
#pragma unroll
        for (int r = 0; r < 8; ++r) {
            int row = rowBase + r + 8 * hlf;       // C/D layout: m = r + 8*half, n = lane&15
            if (row < Nrows) {
                long idx = (long)row * ldc + col;
                float v = acc[c][r] + bv;
                if (accum) v += C[idx];
                if (relu)  v = fmaxf(v, 0.0f);
                C[idx] = v;
            }
        }
    }
}

// ---------- LayerNorm + ReLU (+ optional residual accumulate, optional bf16 copy) ----------
__global__ __launch_bounds__(256) void ln_relu_kernel(
    const float* __restrict__ in, const float* __restrict__ g, const float* __restrict__ be,
    float* __restrict__ res, u16* __restrict__ outb, int ostride, int N)
{
    int row = blockIdx.x * 8 + (threadIdx.x >> 5);
    if (row >= N) return;
    int lane = threadIdx.x & 31;
    const float4 v = *(const float4*)(in + (long)row * 128 + lane * 4);
    float s  = v.x + v.y + v.z + v.w;
    float s2 = v.x * v.x + v.y * v.y + v.z * v.z + v.w * v.w;
#pragma unroll
    for (int off = 16; off >= 1; off >>= 1) {
        s  += __shfl_xor(s,  off, 32);
        s2 += __shfl_xor(s2, off, 32);
    }
    float mu  = s * (1.0f / 128.0f);
    float var = s2 * (1.0f / 128.0f) - mu * mu;
    float inv = rsqrtf(var + 1e-5f);
    float vv[4] = { v.x, v.y, v.z, v.w };
    float o[4];
#pragma unroll
    for (int i = 0; i < 4; ++i) {
        int c = lane * 4 + i;
        o[i] = fmaxf((vv[i] - mu) * inv * g[c] + be[c], 0.0f);
    }
    if (res) {
        float* rp = res + (long)row * 128 + lane * 4;
        float4 r = *(float4*)rp;
        o[0] += r.x; o[1] += r.y; o[2] += r.z; o[3] += r.w;
        float4 w = { o[0], o[1], o[2], o[3] };
        *(float4*)rp = w;
    }
    if (outb) {
        u16* op = outb + (long)row * ostride + lane * 4;
#pragma unroll
        for (int i = 0; i < 4; ++i) op[i] = f2bf(o[i]);
    }
}

// ---------- GCN: self-loop init + edge scatter ----------
__global__ void gcn_init_kernel(const float* __restrict__ xw, const float* __restrict__ dinv,
                                const float* __restrict__ bias, float* __restrict__ out, int N)
{
    int idx = blockIdx.x * 256 + threadIdx.x;
    if (idx < N * 128) {
        int row = idx >> 7, col = idx & 127;
        float di = dinv[row];
        out[idx] = xw[idx] * di * di + bias[col];
    }
}
__global__ __launch_bounds__(256) void gcn_scatter_kernel(
    const float* __restrict__ xw, float* __restrict__ out,
    const int* __restrict__ ei, const float* __restrict__ dinv, int E)
{
    int e = blockIdx.x * 8 + (threadIdx.x >> 5);
    if (e >= E) return;
    int lane = threadIdx.x & 31;
    int s = ei[e], d = ei[E + e];
    float w = dinv[s] * dinv[d];
    const float4 v = *(const float4*)(xw + (long)s * 128 + lane * 4);
    float* o = out + (long)d * 128 + lane * 4;
    atomicAdd(o + 0, v.x * w); atomicAdd(o + 1, v.y * w);
    atomicAdd(o + 2, v.z * w); atomicAdd(o + 3, v.w * w);
}

// ---------- SAGE: mean aggregation ----------
__global__ __launch_bounds__(256) void sage_scatter_kernel(
    const float* __restrict__ h, float* __restrict__ agg, const int* __restrict__ ei, int E)
{
    int e = blockIdx.x * 8 + (threadIdx.x >> 5);
    if (e >= E) return;
    int lane = threadIdx.x & 31;
    int s = ei[e], d = ei[E + e];
    const float4 v = *(const float4*)(h + (long)s * 128 + lane * 4);
    float* o = agg + (long)d * 128 + lane * 4;
    atomicAdd(o + 0, v.x); atomicAdd(o + 1, v.y);
    atomicAdd(o + 2, v.z); atomicAdd(o + 3, v.w);
}
__global__ void sage_div_cvt_kernel(const float* __restrict__ agg, const float* __restrict__ cntinv,
                                    u16* __restrict__ out, int N)
{
    int idx = blockIdx.x * 256 + threadIdx.x;
    if (idx < N * 128) {
        int row = idx >> 7;
        out[idx] = f2bf(agg[idx] * cntinv[row]);
    }
}

// ============================== host ==============================
extern "C" void kernel_launch(void* const* d_in, const int* in_sizes, int n_in,
                              void* d_out, int out_size, void* d_ws, size_t ws_size,
                              hipStream_t stream)
{
    const float* x      = (const float*)d_in[0];
    const int*   ei     = (const int*)  d_in[1];
    const float* W_in1  = (const float*)d_in[2];
    const float* b_in1  = (const float*)d_in[3];
    const float* g_in   = (const float*)d_in[4];
    const float* be_in  = (const float*)d_in[5];
    const float* W_in2  = (const float*)d_in[6];
    const float* b_in2  = (const float*)d_in[7];
    const float* W_gcn  = (const float*)d_in[8];
    const float* b_gcn  = (const float*)d_in[9];
    const float* g0     = (const float*)d_in[10];
    const float* be0    = (const float*)d_in[11];
    const float* W_sl   = (const float*)d_in[12];
    const float* b_sl   = (const float*)d_in[13];
    const float* W_sr   = (const float*)d_in[14];
    const float* g1     = (const float*)d_in[15];
    const float* be1    = (const float*)d_in[16];
    const float* W_jk   = (const float*)d_in[17];
    const float* b_jk   = (const float*)d_in[18];
    const float* W_c1   = (const float*)d_in[19];
    const float* b_c1   = (const float*)d_in[20];
    const float* W_c2   = (const float*)d_in[21];
    const float* b_c2   = (const float*)d_in[22];

    const int N = in_sizes[0] / 128;
    const int E = in_sizes[1] / 2;

    // bump allocator over d_ws
    char* wbase = (char*)d_ws;
    size_t off = 0;
    auto alloc = [&](size_t bytes) -> void* {
        off = (off + 255) & ~(size_t)255;
        void* p = wbase + off;
        off += bytes;
        return p;
    };
    int*   deg_src = (int*)  alloc((size_t)N * 4);
    int*   indeg   = (int*)  alloc((size_t)N * 4);
    int*   hist    = (int*)  alloc((size_t)HB * 4);
    float* med     = (float*)alloc(256);
    float* dinv    = (float*)alloc((size_t)N * 4);
    float* cntinv  = (float*)alloc((size_t)N * 4);
    u16*   xin     = (u16*)  alloc((size_t)N * 160 * 2);
    u16*   hcat    = (u16*)  alloc((size_t)N * 384 * 2);   // [h0 | h1 | h2] bf16
    u16*   abuf    = (u16*)  alloc((size_t)N * 128 * 2);
    float* t0      = (float*)alloc((size_t)N * 128 * 4);
    float* t1      = (float*)alloc((size_t)N * 128 * 4);
    float* h       = (float*)alloc((size_t)N * 128 * 4);
    u16* Wi1b = (u16*)alloc(160 * 128 * 2);
    u16* Wi2b = (u16*)alloc(128 * 128 * 2);
    u16* Wgb  = (u16*)alloc(128 * 128 * 2);
    u16* Wslb = (u16*)alloc(128 * 128 * 2);
    u16* Wsrb = (u16*)alloc(128 * 128 * 2);
    u16* Wjkb = (u16*)alloc(384 * 128 * 2);
    u16* Wc1b = (u16*)alloc(128 * 128 * 2);
    u16* Wc2b = (u16*)alloc(128 * 64 * 2);

    auto cdiv = [](int a, int b) { return (a + b - 1) / b; };
    const int NH = N * 128;

    // ---- graph stats ----
    fill_i32_kernel<<<cdiv(N, 256), 256, 0, stream>>>(deg_src, 0, N);
    fill_i32_kernel<<<cdiv(N, 256), 256, 0, stream>>>(indeg, 0, N);
    fill_i32_kernel<<<cdiv(HB, 256), 256, 0, stream>>>(hist, 0, HB);
    count_deg_kernel<<<cdiv(E, 256), 256, 0, stream>>>(ei, deg_src, indeg, E);
    hist_kernel<<<cdiv(N, 256), 256, 0, stream>>>(deg_src, hist, N);
    median_kernel<<<1, 32, 0, stream>>>(hist, med, (N - 1) / 2);
    invdeg_kernel<<<cdiv(N, 256), 256, 0, stream>>>(indeg, dinv, cntinv, N);

    // ---- input features ----
    float degMean = (float)E / (float)N;
    xin_kernel<<<cdiv(N, 8), 256, 0, stream>>>(x, deg_src, med, xin, N, 1.0f / (degMean + 1e-6f));

    // ---- weights -> bf16 (K padded to multiple of 32) ----
    wconv_kernel<<<cdiv(160 * 128, 256), 256, 0, stream>>>(W_in1, Wi1b, 132, 128, 160);
    wconv_kernel<<<cdiv(128 * 128, 256), 256, 0, stream>>>(W_in2, Wi2b, 128, 128, 128);
    wconv_kernel<<<cdiv(128 * 128, 256), 256, 0, stream>>>(W_gcn, Wgb, 128, 128, 128);
    wconv_kernel<<<cdiv(128 * 128, 256), 256, 0, stream>>>(W_sl, Wslb, 128, 128, 128);
    wconv_kernel<<<cdiv(128 * 128, 256), 256, 0, stream>>>(W_sr, Wsrb, 128, 128, 128);
    wconv_kernel<<<cdiv(384 * 128, 256), 256, 0, stream>>>(W_jk, Wjkb, 384, 128, 384);
    wconv_kernel<<<cdiv(128 * 128, 256), 256, 0, stream>>>(W_c1, Wc1b, 128, 128, 128);
    wconv_kernel<<<cdiv(128 * 64, 256), 256, 0, stream>>>(W_c2, Wc2b, 128, 64, 128);

    const int gemmGrid = cdiv(N, 128);
    auto gemm128 = [&](const u16* A, int lda, const u16* Bw, int K,
                       const float* bias, float* C, int flags) {
        wmma_gemm_kernel<8><<<gemmGrid, 256, (size_t)K * 128 * 2, stream>>>(
            A, lda, Bw, bias, C, 128, N, K, flags);
    };

    // ---- input MLP ----
    gemm128(xin, 160, Wi1b, 160, b_in1, t0, 0);                       // t0 = xin@W_in1 + b
    ln_relu_kernel<<<cdiv(N, 8), 256, 0, stream>>>(t0, g_in, be_in, nullptr, abuf, 128, N);
    gemm128(abuf, 128, Wi2b, 128, b_in2, h, 0);                       // h0 = relu(LN)@W_in2 + b
    cvt_rows_kernel<<<cdiv(NH, 256), 256, 0, stream>>>(h, hcat + 0, 384, N);

    // ---- layer 0: GCN + LN + ReLU + residual ----
    gemm128(hcat + 0, 384, Wgb, 128, nullptr, t0, 0);                 // t0 = h@W_gcn
    gcn_init_kernel<<<cdiv(NH, 256), 256, 0, stream>>>(t0, dinv, b_gcn, t1, N);
    gcn_scatter_kernel<<<cdiv(E, 8), 256, 0, stream>>>(t0, t1, ei, dinv, E);
    ln_relu_kernel<<<cdiv(N, 8), 256, 0, stream>>>(t1, g0, be0, h, hcat + 128, 384, N);

    // ---- layer 1: SAGE + LN + ReLU + residual ----
    fill_f32_kernel<<<cdiv(NH, 256), 256, 0, stream>>>(t0, 0.0f, NH);
    sage_scatter_kernel<<<cdiv(E, 8), 256, 0, stream>>>(h, t0, ei, E);
    sage_div_cvt_kernel<<<cdiv(NH, 256), 256, 0, stream>>>(t0, cntinv, abuf, N);
    gemm128(abuf, 128, Wslb, 128, b_sl, t1, 0);                       // t1 = mean_agg@W_sl + b
    gemm128(hcat + 128, 384, Wsrb, 128, nullptr, t1, 1);              // t1 += h@W_sr
    ln_relu_kernel<<<cdiv(N, 8), 256, 0, stream>>>(t1, g1, be1, h, hcat + 256, 384, N);

    // ---- JK projection + classifier ----
    gemm128(hcat, 384, Wjkb, 384, b_jk, t0, 0);                       // z = concat@W_jk + b
    cvt_rows_kernel<<<cdiv(NH, 256), 256, 0, stream>>>(t0, abuf, 128, N);
    gemm128(abuf, 128, Wc1b, 128, b_c1, t1, 2);                       // relu fused
    cvt_rows_kernel<<<cdiv(NH, 256), 256, 0, stream>>>(t1, abuf, 128, N);
    wmma_gemm_kernel<4><<<gemmGrid, 256, (size_t)128 * 64 * 2, stream>>>(
        abuf, 128, Wc2b, b_c2, (float*)d_out, 64, N, 128, 0);         // [N,64] output
}
